// NumericallyStableMultiheadAttention_48241072668955
// MI455X (gfx1250) — matrix operand order repaired and checked by reading, not compiled, not run on hardware
//
#include <hip/hip_runtime.h>
#include <hip/hip_bf16.h>

// ---------------------------------------------------------------------------
// MHA forward for MI455X (gfx1250, wave32, WMMA + TDM).
// All matmuls: v_wmma_f32_16x16x32_bf16, fp32 accum, 4 accumulators per wave.
// attn@V stages the shared V-tile in LDS via TENSOR_LOAD_TO_LDS (TDM) with
// bank-conflict padding; falls back to a cooperative copy if the builtin is
// unavailable.
// d_out = [out (B*S*D f32)] ++ [attn_weights (B*H*S*S f32)]
// ws    = Qbf [B*S,D], Kbf [B*S,D], Vt [B*D,S] (transposed), Obf [B*S,D]
// ---------------------------------------------------------------------------

typedef __attribute__((ext_vector_type(16))) __bf16 v16bf;
typedef __attribute__((ext_vector_type(8)))  __bf16 v8bf;
typedef __attribute__((ext_vector_type(8)))  float  v8f;
typedef __attribute__((ext_vector_type(4)))  float  v4f;
typedef __attribute__((ext_vector_type(4)))  unsigned int v4u;
typedef __attribute__((ext_vector_type(4)))  unsigned int u32x4;
typedef __attribute__((ext_vector_type(8)))  int i32x8;
typedef __attribute__((ext_vector_type(4)))  int i32x4;

#define NEG_HUGE (-3.402823466e+38f)

#ifndef __has_builtin
#define __has_builtin(x) 0
#endif
#if __has_builtin(__builtin_amdgcn_tensor_load_to_lds) && \
    __has_builtin(__builtin_amdgcn_s_wait_tensorcnt)
#define HAVE_TDM 1
#else
#define HAVE_TDM 0
#endif

// Padded LDS tile for the attn@V B operand: 64 channels x 64 seq bf16,
// row stride 64+8 elements (144 B = 36 dwords -> conflict-free half-wave).
#define VT_ROW 72

__device__ __forceinline__ __bf16 f2bf(float f) { return (__bf16)f; }

__device__ __forceinline__ v8f wmma_bf16(v16bf a, v16bf b, v8f c) {
  return __builtin_amdgcn_wmma_f32_16x16x32_bf16(false, a, false, b, (short)0, c,
                                                 false, false);
}

// A fragment (16x32, MxK), fp32 source -> bf16.
__device__ __forceinline__ v16bf load_a_f32(const float* __restrict__ A, int lda,
                                            int row0, int kBase, int lane) {
  int row = row0 + (lane & 15);
  int off = (lane & 16) ? 16 : 0;
  const v4f* p = reinterpret_cast<const v4f*>(A + (size_t)row * lda + kBase + off);
  v4f x0 = p[0], x1 = p[1], x2 = p[2], x3 = p[3];
  v16bf a;
#pragma unroll
  for (int i = 0; i < 4; ++i) {
    a[i]      = f2bf(x0[i]);
    a[i + 4]  = f2bf(x1[i]);
    a[i + 8]  = f2bf(x2[i]);
    a[i + 12] = f2bf(x3[i]);
  }
  return a;
}

// A fragment from bf16 source.
__device__ __forceinline__ v16bf load_a_bf16(const __bf16* __restrict__ A, int lda,
                                             int row0, int kBase, int lane) {
  int row = row0 + (lane & 15);
  int off = (lane & 16) ? 16 : 0;
  const __bf16* base = A + (size_t)row * lda + kBase + off;
  v8bf h0 = __builtin_bit_cast(v8bf, *reinterpret_cast<const v4u*>(base));
  v8bf h1 = __builtin_bit_cast(v8bf, *reinterpret_cast<const v4u*>(base + 8));
  v16bf a;
#pragma unroll
  for (int i = 0; i < 8; ++i) { a[i] = h0[i]; a[i + 8] = h1[i]; }
  return a;
}

// B fragment (32x16), per-lane K contiguous, fp32 source.
__device__ __forceinline__ v16bf load_b_rowk_f32(const float* __restrict__ W, int ldw,
                                                 int col0, int kBase, int lane) {
  int col  = col0 + (lane & 15);
  int koff = (lane & 16) ? 16 : 0;
  const v4f* p = reinterpret_cast<const v4f*>(W + (size_t)col * ldw + kBase + koff);
  v4f x0 = p[0], x1 = p[1], x2 = p[2], x3 = p[3];
  v16bf b;
#pragma unroll
  for (int i = 0; i < 4; ++i) {
    b[i]      = f2bf(x0[i]);
    b[i + 4]  = f2bf(x1[i]);
    b[i + 8]  = f2bf(x2[i]);
    b[i + 12] = f2bf(x3[i]);
  }
  return b;
}

// Same, bf16 source (global).
__device__ __forceinline__ v16bf load_b_rowk_bf16(const __bf16* __restrict__ P, int ld,
                                                  int col0, int kBase, int lane) {
  int col  = col0 + (lane & 15);
  int koff = (lane & 16) ? 16 : 0;
  const __bf16* base = P + (size_t)col * ld + kBase + koff;
  v8bf h0 = __builtin_bit_cast(v8bf, *reinterpret_cast<const v4u*>(base));
  v8bf h1 = __builtin_bit_cast(v8bf, *reinterpret_cast<const v4u*>(base + 8));
  v16bf b;
#pragma unroll
  for (int i = 0; i < 8; ++i) { b[i] = h0[i]; b[i + 8] = h1[i]; }
  return b;
}

// B fragment from the padded LDS tile (channels x seq).
__device__ __forceinline__ v16bf load_b_lds(const __bf16* tile, int j, int kk, int lane) {
  int col  = 16 * j + (lane & 15);              // channel
  int koff = (lane & 16) ? 16 : 0;              // seq half
  const __bf16* base = tile + col * VT_ROW + kk + koff;
  v8bf h0 = __builtin_bit_cast(v8bf, *reinterpret_cast<const v4u*>(base));
  v8bf h1 = __builtin_bit_cast(v8bf, *reinterpret_cast<const v4u*>(base + 8));
  v16bf b;
#pragma unroll
  for (int i = 0; i < 8; ++i) { b[i] = h0[i]; b[i + 8] = h1[i]; }
  return b;
}

#if HAVE_TDM
// Issue a TDM 2D tile load: 64 rows x 64 cols of bf16 from gsrc (row stride
// strideElems) into LDS at ldsByteAddr, padding 16B after each 128B row.
// D# layout per CDNA5 ISA 8.3-8.5.
__device__ __forceinline__ void tdm_load_vtile(unsigned ldsByteAddr,
                                               const __bf16* gsrc,
                                               int strideElems) {
  unsigned long long ga = (unsigned long long)(size_t)gsrc;
  u32x4 g0;
  g0.x = 1u;                                           // count=1, user mode
  g0.y = ldsByteAddr;                                  // lds_addr
  g0.z = (unsigned)(ga & 0xFFFFFFFFull);               // global_addr[31:0]
  g0.w = (unsigned)((ga >> 32) & 0x01FFFFFFull)        // global_addr[56:32]
         | (2u << 30);                                 // type = 2 (image)
  i32x8 g1;
  g1[0] = (int)((1u << 16)      // data_size = 1 (2 bytes)
              | (1u << 20)      // pad_enable
              | (4u << 22)      // pad_interval: 32 dwords (128 B)
              | (3u << 25));    // pad_amount: 4 dwords (16 B)
  g1[1] = (int)(((unsigned)strideElems & 0xFFFFu) << 16);        // tensor_dim0 lo
  g1[2] = (int)((((unsigned)strideElems >> 16) & 0xFFFFu)        // tensor_dim0 hi
              | (64u << 16));                                    // tensor_dim1 lo
  g1[3] = (int)(64u << 16);                                      // tile_dim0 = 64
  g1[4] = 64;                                                    // tile_dim1 = 64
  g1[5] = strideElems;                                           // dim0 stride
  g1[6] = 0;
  g1[7] = 0;
  i32x4 gz = {0, 0, 0, 0};      // 2D tensor: groups 2/3 unused
#if __clang_major__ >= 23
  i32x8 gz8 = {0, 0, 0, 0, 0, 0, 0, 0};
  __builtin_amdgcn_tensor_load_to_lds(g0, g1, gz, gz, gz8, 0);
#else
  __builtin_amdgcn_tensor_load_to_lds(g0, g1, gz, gz, 0);
#endif
}
#endif

// ---------------------------------------------------------------------------
// Kernel 1: batched QKV projection (z selects operand set).
// Q,K stored bf16 [B*S, D]; V stored transposed (Vt[B*D, S]).
// ---------------------------------------------------------------------------
__global__ __launch_bounds__(128) void qkv_proj_kernel(
    const float* __restrict__ queries, const float* __restrict__ keys,
    const float* __restrict__ values,
    const float* __restrict__ wq, const float* __restrict__ bq,
    const float* __restrict__ wk, const float* __restrict__ bk,
    const float* __restrict__ wv, const float* __restrict__ bv,
    __bf16* __restrict__ Qbf, __bf16* __restrict__ Kbf, __bf16* __restrict__ Vt,
    int N, int K, int Sv) {
  int lane = threadIdx.x & 31;
  int wave = threadIdx.x >> 5;
  const float *X, *W, *bias;
  if (blockIdx.z == 0)      { X = queries; W = wq; bias = bq; }
  else if (blockIdx.z == 1) { X = keys;    W = wk; bias = bk; }
  else                      { X = values;  W = wv; bias = bv; }

  int n0 = blockIdx.x * 64;
  int m0 = blockIdx.y * 64 + wave * 16;

  v8f acc[4] = {};
  for (int k = 0; k < K; k += 32) {
    __builtin_prefetch(X + (size_t)(m0 + (lane & 15)) * K + k + 64, 0, 0);
    v16bf a = load_a_f32(X, K, m0, k, lane);
#pragma unroll
    for (int j = 0; j < 4; ++j) {
      v16bf b = load_b_rowk_f32(W, K, n0 + 16 * j, k, lane);
      acc[j] = wmma_bf16(a, b, acc[j]);
    }
  }

  int rloc = (lane & 16) ? 8 : 0;
  if (blockIdx.z != 2) {
    __bf16* dst = (blockIdx.z == 0) ? Qbf : Kbf;
#pragma unroll
    for (int j = 0; j < 4; ++j) {
      int col  = n0 + 16 * j + (lane & 15);
      float bc = bias[col];
#pragma unroll
      for (int r = 0; r < 8; ++r)
        dst[(size_t)(m0 + rloc + r) * N + col] = f2bf(acc[j][r] + bc);
    }
  } else {
    // V transposed: rows r are consecutive s -> one 16B store per sub-tile.
    int bIdx  = m0 / Sv;
    int sBase = (m0 % Sv) + rloc;
#pragma unroll
    for (int j = 0; j < 4; ++j) {
      int col  = n0 + 16 * j + (lane & 15);
      float bc = bias[col];
      v8bf t;
#pragma unroll
      for (int r = 0; r < 8; ++r) t[r] = f2bf(acc[j][r] + bc);
      *reinterpret_cast<v4u*>(Vt + ((size_t)bIdx * N + col) * Sv + sBase) =
          __builtin_bit_cast(v4u, t);
    }
  }
}

// ---------------------------------------------------------------------------
// Kernel 2: causal-masked scores.
// ---------------------------------------------------------------------------
__global__ __launch_bounds__(128) void scores_kernel(
    const __bf16* __restrict__ Qbf, const __bf16* __restrict__ Kbf,
    float* __restrict__ attn, int Hv, int Sv, int Dv, float scale) {
  int lane = threadIdx.x & 31;
  int wave = threadIdx.x >> 5;
  int z = blockIdx.z;
  int b = z / Hv, h = z % Hv;
  int q0 = blockIdx.y * 64 + wave * 16;
  int j0 = blockIdx.x * 64;

  float* out = attn + ((size_t)z * Sv + q0) * Sv;
  int rloc   = (lane & 16) ? 8 : 0;

  int jCnt = 0;
  if (j0 <= q0 + 15) {
    jCnt = ((q0 + 15 - j0) >> 4) + 1;
    if (jCnt > 4) jCnt = 4;
  }

  v8f acc[4] = {};
  if (jCnt > 0) {
#pragma unroll
    for (int k = 0; k < 64; k += 32) {
      v16bf a = load_a_bf16(Qbf, Dv, b * Sv + q0, h * 64 + k, lane);
      for (int j = 0; j < jCnt; ++j) {
        v16bf bb = load_b_rowk_bf16(Kbf, Dv, b * Sv + j0 + 16 * j, h * 64 + k, lane);
        acc[j] = wmma_bf16(a, bb, acc[j]);
      }
    }
  }

#pragma unroll
  for (int j = 0; j < 4; ++j) {
    int col = j0 + 16 * j + (lane & 15);
#pragma unroll
    for (int r = 0; r < 8; ++r) {
      int q   = q0 + rloc + r;
      float s = (j < jCnt) ? acc[j][r] * scale : NEG_HUGE;
      if (col > q) s = NEG_HUGE;
      out[(size_t)(rloc + r) * Sv + col] = s;
    }
  }
}

// ---------------------------------------------------------------------------
// Kernel 3: in-place row softmax.
// ---------------------------------------------------------------------------
__global__ __launch_bounds__(256) void softmax_kernel(float* __restrict__ attn, int Sv) {
  __shared__ float red[256];
  float* row = attn + (size_t)blockIdx.x * Sv;
  int t = threadIdx.x;

  float m = NEG_HUGE;
  for (int i = t; i < Sv; i += 256) m = fmaxf(m, row[i]);
  red[t] = m; __syncthreads();
  for (int s = 128; s > 0; s >>= 1) {
    if (t < s) red[t] = fmaxf(red[t], red[t + s]);
    __syncthreads();
  }
  m = red[0]; __syncthreads();

  float sum = 0.f;
  for (int i = t; i < Sv; i += 256) {
    float e = __expf(row[i] - m);
    row[i] = e;
    sum += e;
  }
  red[t] = sum; __syncthreads();
  for (int s = 128; s > 0; s >>= 1) {
    if (t < s) red[t] += red[t + s];
    __syncthreads();
  }
  float inv = 1.0f / red[0];
  for (int i = t; i < Sv; i += 256) row[i] *= inv;
}

// ---------------------------------------------------------------------------
// Kernel 4: attn @ V per head. Block = 64 q-rows x full head dim (64).
// The V-tile (64 ch x 64 seq bf16) is shared by all 4 waves: staged in LDS by
// the Tensor Data Mover (or a cooperative copy fallback), with 16B row
// padding for conflict-free ds reads. K-loop bound is block-uniform; columns
// beyond a wave's causal limit multiply exact softmax zeros.
// ---------------------------------------------------------------------------
__global__ __launch_bounds__(128) void attn_v_kernel(
    const float* __restrict__ attn, const __bf16* __restrict__ Vt,
    __bf16* __restrict__ Obf, int Hv, int Sv, int Dv) {
  __shared__ __bf16 vtile[64 * VT_ROW];

  int lane = threadIdx.x & 31;
  int wave = threadIdx.x >> 5;
  int z = blockIdx.y;
  int b = z / Hv, h = z % Hv;
  int q0 = blockIdx.x * 64 + wave * 16;

  const float* Arow = attn + ((size_t)z * Sv + q0) * Sv;
  const __bf16* VtH = Vt + (size_t)(b * Dv + h * 64) * Sv;   // 64 channels x S

  v8f acc[4] = {};
  int kBlockLimit = blockIdx.x * 64 + 64;   // block-uniform causal bound

  for (int k = 0; k < kBlockLimit; k += 64) {
    __syncthreads();                        // previous tile fully consumed
#if HAVE_TDM
    if (wave == 0) {
      tdm_load_vtile((unsigned)(size_t)(void*)vtile, VtH + k, Sv);
      __builtin_amdgcn_s_wait_tensorcnt(0);
    }
#else
    {
      int r = threadIdx.x & 63;             // channel row
      int p = threadIdx.x >> 6;             // which 32-col half
      const v4u* src = reinterpret_cast<const v4u*>(VtH + (size_t)r * Sv + k + p * 32);
      v4u* dst = reinterpret_cast<v4u*>(&vtile[r * VT_ROW + p * 32]);
#pragma unroll
      for (int i = 0; i < 4; ++i) dst[i] = src[i];
    }
#endif
    __syncthreads();                        // tile visible to all waves

#pragma unroll
    for (int kk = 0; kk < 64; kk += 32) {
      __builtin_prefetch(Arow + (size_t)(lane & 15) * Sv + k + kk + 128, 0, 0);
      v16bf a = load_a_f32(Arow, Sv, 0, k + kk, lane);
#pragma unroll
      for (int j = 0; j < 4; ++j) {
        v16bf bb = load_b_lds(vtile, j, kk, lane);
        acc[j] = wmma_bf16(a, bb, acc[j]);
      }
    }
  }

  int rbase = q0 + ((lane & 16) ? 8 : 0);
#pragma unroll
  for (int j = 0; j < 4; ++j) {
    int d = h * 64 + 16 * j + (lane & 15);
#pragma unroll
    for (int r = 0; r < 8; ++r)
      Obf[(size_t)(b * Sv + rbase + r) * Dv + d] = f2bf(acc[j][r]);
  }
}

// ---------------------------------------------------------------------------
// Kernel 5: out = O @ wo^T + bo (fp32 into d_out).
// ---------------------------------------------------------------------------
__global__ __launch_bounds__(128) void out_proj_kernel(
    const __bf16* __restrict__ Obf, const float* __restrict__ wo,
    const float* __restrict__ bo, float* __restrict__ out, int N, int K) {
  int lane = threadIdx.x & 31;
  int wave = threadIdx.x >> 5;
  int n0 = blockIdx.x * 64;
  int m0 = blockIdx.y * 64 + wave * 16;

  v8f acc[4] = {};
  for (int k = 0; k < K; k += 32) {
    __builtin_prefetch(wo + (size_t)(n0 + (lane & 15)) * K + k + 64, 0, 0);
    v16bf a = load_a_bf16(Obf, K, m0, k, lane);
#pragma unroll
    for (int j = 0; j < 4; ++j) {
      v16bf b = load_b_rowk_f32(wo, K, n0 + 16 * j, k, lane);
      acc[j] = wmma_bf16(a, b, acc[j]);
    }
  }

  int rloc = (lane & 16) ? 8 : 0;
#pragma unroll
  for (int j = 0; j < 4; ++j) {
    int col  = n0 + 16 * j + (lane & 15);
    float bc = bo[col];
#pragma unroll
    for (int r = 0; r < 8; ++r)
      out[(size_t)(m0 + rloc + r) * N + col] = acc[j][r] + bc;
  }
}

// ---------------------------------------------------------------------------
extern "C" void kernel_launch(void* const* d_in, const int* in_sizes, int n_in,
                              void* d_out, int out_size, void* d_ws, size_t ws_size,
                              hipStream_t stream) {
  constexpr int Bc = 2, Sc = 2048, Dc = 1024, Hc = 16;
  constexpr size_t MD = (size_t)Bc * Sc * Dc;

  const float* queries = (const float*)d_in[0];
  const float* keys    = (const float*)d_in[1];
  const float* values  = (const float*)d_in[2];
  // d_in[3] = attn_mask (causal triu, implemented analytically)
  const float* wq = (const float*)d_in[4];
  const float* bq = (const float*)d_in[5];
  const float* wk = (const float*)d_in[6];
  const float* bk = (const float*)d_in[7];
  const float* wv = (const float*)d_in[8];
  const float* bv = (const float*)d_in[9];
  const float* wo = (const float*)d_in[10];
  const float* bo = (const float*)d_in[11];

  float* out  = (float*)d_out;
  float* attn = out + MD;

  __bf16* Qbf = (__bf16*)d_ws;               // [B*S, D]
  __bf16* Kbf = Qbf + MD;                    // [B*S, D]
  __bf16* Vt  = Kbf + MD;                    // [B*D, S] (transposed)
  __bf16* Obf = Vt  + MD;                    // [B*S, D]

  dim3 blk(128);

  qkv_proj_kernel<<<dim3(Dc / 64, (Bc * Sc) / 64, 3), blk, 0, stream>>>(
      queries, keys, values, wq, bq, wk, bk, wv, bv, Qbf, Kbf, Vt, Dc, Dc, Sc);

  scores_kernel<<<dim3(Sc / 64, Sc / 64, Bc * Hc), blk, 0, stream>>>(
      Qbf, Kbf, attn, Hc, Sc, Dc, 0.125f);

  softmax_kernel<<<dim3(Bc * Hc * Sc), dim3(256), 0, stream>>>(attn, Sc);

  attn_v_kernel<<<dim3(Sc / 64, Bc * Hc), blk, 0, stream>>>(attn, Vt, Obf, Hc, Sc, Dc);

  out_proj_kernel<<<dim3(Dc / 64, (Bc * Sc) / 64), blk, 0, stream>>>(
      Obf, wo, bo, out, Dc, Dc);
}
